// MultiHeadAttention_54133767799241
// MI455X (gfx1250) — compile-verified, matrix operand order checked by reading, not run on hardware
//
#include <hip/hip_runtime.h>
#include <cstdint>
#include <cstddef>

typedef __attribute__((ext_vector_type(16))) _Float16 v16h;
typedef __attribute__((ext_vector_type(8)))  _Float16 v8h;
typedef __attribute__((ext_vector_type(8)))  float    v8f;

#define DEV __device__ __forceinline__

// Problem constants (match reference)
constexpr int BB  = 4;
constexpr int TT  = 2048;
constexpr int HH  = 16;
constexpr int DD  = 64;
constexpr int CC  = 1024;   // n_embd
constexpr int IND = 132;
constexpr int INP = 160;    // 132 padded to mult of 32
constexpr int OD  = 136;    // out dim
constexpr int ODP = 256;    // padded N for proj GEMM (2 * 128 block tiles)
constexpr int MM  = BB * TT; // 8192

// sqrt(n_embd) * log2(e): attention logits are kept in base-2 domain so the
// softmax exponentials are bare v_exp_f32 (no per-element mul by log2e).
constexpr float QSCALE = 32.0f * 1.44269504088896340736f;

// ---------------------------------------------------------------------------
// CDNA5 async global->LDS copy (ASYNCcnt tracked). LDS dest address is the
// wave-relative offset = low 32 bits of the generic shared pointer.
DEV void async_cp_b128(const _Float16* gsrc, const _Float16* ldst) {
    uint32_t loff = (uint32_t)(uintptr_t)ldst;
    uint64_t gaddr = (uint64_t)(uintptr_t)gsrc;
    asm volatile("global_load_async_to_lds_b128 %0, %1, off"
                 :: "v"(loff), "v"(gaddr) : "memory");
}
DEV void wait_async0() { asm volatile("s_wait_asynccnt 0" ::: "memory"); }

DEV float exp2_fast(float x) { return __builtin_amdgcn_exp2f(x); }

// shuffle float through int bits (avoids float canonicalization on the
// bpermute result)
DEV float shflx_f(float v, int off) {
    return __int_as_float(__shfl_xor(__float_as_int(v), off, 32));
}

// ---------------------------------------------------------------------------
// Fragment loader: A-frag (16x32 f16, M rows) and B-frag (32x16, N cols from
// Bt row-major [N,K]) share the same lane layout.
//   lane 0-15 : row = lane,    halves 0..7 -> K {0..7},  halves 8..15 -> K {16..23}
//   lane16-31 : row = lane-16, halves 0..7 -> K {8..15}, halves 8..15 -> K {24..31}
DEV v16h frag16(const _Float16* p, int g) {
    v8h lo = *(const v8h*)(p + 8 * g);
    v8h hi = *(const v8h*)(p + 16 + 8 * g);
    v16h r;
#pragma unroll
    for (int i = 0; i < 8; ++i) { r[i] = lo[i]; r[i + 8] = hi[i]; }
    return r;
}

DEV v8f wmma_f16(v16h a, v16h b, v8f c) {
    return __builtin_amdgcn_wmma_f32_16x16x32_f16(false, a, false, b,
                                                  (short)0, c, false, false);
}

// ---------------------------------------------------------------------------
// Prep kernels: f32 -> f16 with padding / transpose.

__global__ void prep_x(const float* __restrict__ x, _Float16* __restrict__ xh) {
    int idx = blockIdx.x * blockDim.x + threadIdx.x;
    if (idx >= MM * INP) return;
    int t = idx / INP, c = idx - t * INP;
    xh[idx] = (c < IND) ? (_Float16)x[(size_t)t * IND + c] : (_Float16)0.f;
}

__global__ void prep_wembT(const float* __restrict__ w, _Float16* __restrict__ o) {
    int idx = blockIdx.x * blockDim.x + threadIdx.x;
    if (idx >= CC * INP) return;
    int n = idx / INP, k = idx - n * INP;
    o[idx] = (k < IND) ? (_Float16)w[(size_t)k * CC + n] : (_Float16)0.f;
}

// w_q/w_k/w_v: [H, C, D] f32 -> [H*D, C] f16 (row n = h*64+d holds column (h,d))
__global__ void prep_wqkv(const float* __restrict__ wq, const float* __restrict__ wk,
                          const float* __restrict__ wv,
                          _Float16* __restrict__ qT, _Float16* __restrict__ kT,
                          _Float16* __restrict__ vT) {
    int idx = blockIdx.x * blockDim.x + threadIdx.x;
    if (idx >= CC * CC) return;
    int n = idx >> 10, c = idx & 1023;
    int h = n >> 6, d = n & 63;
    size_t src = ((size_t)h * CC + c) * DD + d;
    qT[idx] = (_Float16)wq[src];
    kT[idx] = (_Float16)wk[src];
    vT[idx] = (_Float16)wv[src];
}

// w_proj [C, 136] f32 -> [256, C] f16 zero-padded rows >= 136
__global__ void prep_wprojT(const float* __restrict__ w, _Float16* __restrict__ o) {
    int idx = blockIdx.x * blockDim.x + threadIdx.x;
    if (idx >= ODP * CC) return;
    int n = idx >> 10, c = idx & 1023;
    o[idx] = (n < OD) ? (_Float16)w[(size_t)c * OD + n] : (_Float16)0.f;
}

// ---------------------------------------------------------------------------
// Generic f16 WMMA GEMM: C = alpha * A[M,K] @ Bt[N,K]^T (+ bias)
// Block tile 128x128, 8 waves (4x2), wave tile 32x64, BK=32.
// Async global->LDS staging, ping-pong double buffer, one barrier per K step.
template <bool F16OUT, bool BIAS>
__global__ __launch_bounds__(256) void gemm_f16w(
    const _Float16* __restrict__ A, const _Float16* __restrict__ Bt,
    const float* __restrict__ bias, _Float16* __restrict__ Cf16,
    float* __restrict__ Cf32, int K, int ldc, int nvalid, float alpha) {

    __shared__ _Float16 As[2 * 128 * 40];
    __shared__ _Float16 Bs[2 * 128 * 40];

    const int tid = threadIdx.x, lane = tid & 31;
    const int w = tid >> 5, wm = w >> 1, wn = w & 1;
    const int g = lane >> 4, ml = lane & 15;
    const int m0 = blockIdx.x * 128, n0 = blockIdx.y * 128;
    const int nk = K >> 5;

    // per-thread staging source/dest (2 chunks of A, 2 of B per tile)
    const int r0 = tid >> 2, c80 = (tid & 3) * 8;
    const int r1 = (tid + 256) >> 2, c81 = ((tid + 256) & 3) * 8;
    const _Float16* gA0 = A + (size_t)(m0 + r0) * K + c80;
    const _Float16* gA1 = A + (size_t)(m0 + r1) * K + c81;
    const _Float16* gB0 = Bt + (size_t)(n0 + r0) * K + c80;
    const _Float16* gB1 = Bt + (size_t)(n0 + r1) * K + c81;

    auto issue = [&](int i, int stage) {
        _Float16* as = As + stage * (128 * 40);
        _Float16* bs = Bs + stage * (128 * 40);
        async_cp_b128(gA0 + i * 32, as + r0 * 40 + c80);
        async_cp_b128(gA1 + i * 32, as + r1 * 40 + c81);
        async_cp_b128(gB0 + i * 32, bs + r0 * 40 + c80);
        async_cp_b128(gB1 + i * 32, bs + r1 * 40 + c81);
    };

    v8f acc[2][4];
#pragma unroll
    for (int i = 0; i < 2; ++i)
#pragma unroll
        for (int j = 0; j < 4; ++j)
#pragma unroll
            for (int e = 0; e < 8; ++e) acc[i][j][e] = 0.f;

    issue(0, 0);
    wait_async0();
    __syncthreads();

    for (int i = 0; i < nk; ++i) {
        const int cur = i & 1;
        if (i + 1 < nk) issue(i + 1, cur ^ 1);

        const _Float16* as = As + cur * (128 * 40);
        const _Float16* bs = Bs + cur * (128 * 40);
        v16h af[2], bf[4];
#pragma unroll
        for (int mf = 0; mf < 2; ++mf)
            af[mf] = frag16(as + (wm * 32 + mf * 16 + ml) * 40, g);
#pragma unroll
        for (int nf = 0; nf < 4; ++nf)
            bf[nf] = frag16(bs + (wn * 64 + nf * 16 + ml) * 40, g);
#pragma unroll
        for (int mf = 0; mf < 2; ++mf)
#pragma unroll
            for (int nf = 0; nf < 4; ++nf)
                acc[mf][nf] = wmma_f16(af[mf], bf[nf], acc[mf][nf]);

        wait_async0();
        __syncthreads();
    }

#pragma unroll
    for (int mf = 0; mf < 2; ++mf)
#pragma unroll
        for (int nf = 0; nf < 4; ++nf) {
            const int col = n0 + wn * 64 + nf * 16 + ml;
            float bv = 0.f;
            if constexpr (BIAS) {
                if (F16OUT || col < nvalid) bv = bias[col];
            }
#pragma unroll
            for (int r = 0; r < 8; ++r) {
                const int row = m0 + wm * 32 + mf * 16 + r + 8 * g;
                float v = acc[mf][nf][r] * alpha + bv;
                if constexpr (F16OUT) {
                    Cf16[(size_t)row * ldc + col] = (_Float16)v;
                } else {
                    if (col < nvalid) Cf32[(size_t)row * ldc + col] = v;
                }
            }
        }
}

// ---------------------------------------------------------------------------
// v_all [B*T, C] f16 -> vT_all [B,H,D,T] f16 (64x64 LDS tiles)
__global__ __launch_bounds__(256) void transpose_v(const _Float16* __restrict__ v,
                                                   _Float16* __restrict__ vT) {
    __shared__ _Float16 t[64 * 72];
    const int tid = threadIdx.x;
    const int si = blockIdx.x & 31;
    const int bh = blockIdx.x >> 5;
    const int h = bh & 15, b = bh >> 4;
#pragma unroll
    for (int j = 0; j < 2; ++j) {
        int ci = tid + 256 * j;
        int r = ci >> 3, c8 = (ci & 7) * 8;
        *(v8h*)(t + r * 72 + c8) =
            *(const v8h*)(v + (size_t)(b * TT + si * 64 + r) * CC + h * DD + c8);
    }
    __syncthreads();
#pragma unroll
    for (int j = 0; j < 2; ++j) {
        int ci = tid + 256 * j;
        int d = ci >> 3, s8 = (ci & 7) * 8;
        v8h o;
#pragma unroll
        for (int i = 0; i < 8; ++i) o[i] = t[(s8 + i) * 72 + d];
        *(v8h*)(vT + ((size_t)(b * HH + h) * DD + d) * TT + si * 64 + s8) = o;
    }
}

// ---------------------------------------------------------------------------
// Flash attention: block = 8 waves, each wave owns a private 16-row Q tile of
// one (b,h). K/Vt 64x64 tiles async-staged in ping-pong LDS buffers.
// Q pre-scaled by 32*log2(e) in the Q GEMM; all exponentials are exp2.
// Row-sums of P are computed with an all-ones B-fragment WMMA (no cross-lane
// reduction needed); only the row-max uses shuffle reductions.
__global__ __launch_bounds__(256) void attn(const _Float16* __restrict__ q,
                                            const _Float16* __restrict__ k,
                                            const _Float16* __restrict__ vT,
                                            _Float16* __restrict__ o) {
    // 2 stages x (K tile + Vt tile) + 8 wave-private P bounce buffers
    __shared__ _Float16 sm[2 * (2 * 64 * 72) + 8 * 16 * 72];

    const int tid = threadIdx.x, lane = tid & 31, w = tid >> 5;
    const int g = lane >> 4, ml = lane & 15;
    _Float16* pb = sm + 2 * (2 * 64 * 72) + w * 16 * 72;

    const int qblk = blockIdx.x & 15;
    const int bh = blockIdx.x >> 4;
    const int h = bh & 15, b = bh >> 4;
    const int t0 = qblk * 128 + w * 16;

    // per-thread staging addresses (2 chunks of K, 2 of Vt per tile)
    const int r0 = tid >> 3, c80 = (tid & 7) * 8;
    const int r1 = (tid + 256) >> 3, c81 = ((tid + 256) & 7) * 8;
    const _Float16* gK0 = k + (size_t)(b * TT + r0) * CC + h * DD + c80;
    const _Float16* gK1 = k + (size_t)(b * TT + r1) * CC + h * DD + c81;
    const _Float16* gV0 = vT + ((size_t)(b * HH + h) * DD + r0) * TT + c80;
    const _Float16* gV1 = vT + ((size_t)(b * HH + h) * DD + r1) * TT + c81;

    auto issue = [&](int i, int stage) {
        _Float16* kb = sm + stage * (2 * 64 * 72);
        _Float16* vb = kb + 64 * 72;
        async_cp_b128(gK0 + (size_t)(i * 64) * CC, kb + r0 * 72 + c80);
        async_cp_b128(gK1 + (size_t)(i * 64) * CC, kb + r1 * 72 + c81);
        async_cp_b128(gV0 + i * 64, vb + r0 * 72 + c80);
        async_cp_b128(gV1 + i * 64, vb + r1 * 72 + c81);
    };

    v16h qa[2];
#pragma unroll
    for (int kk = 0; kk < 2; ++kk)
        qa[kk] = frag16(q + (size_t)(b * TT + t0 + ml) * CC + h * DD + kk * 32, g);

    v16h ones;
#pragma unroll
    for (int i = 0; i < 16; ++i) ones[i] = (_Float16)1.f;

    float mi[8], li[8];
    v8f oacc[4];
#pragma unroll
    for (int r = 0; r < 8; ++r) { mi[r] = -3.0e38f; li[r] = 0.f; }
#pragma unroll
    for (int nf = 0; nf < 4; ++nf)
#pragma unroll
        for (int e = 0; e < 8; ++e) oacc[nf][e] = 0.f;

    issue(0, 0);
    wait_async0();
    __syncthreads();

    constexpr int NS = TT / 64;
    for (int i = 0; i < NS; ++i) {
        const int cur = i & 1;
        if (i + 1 < NS) issue(i + 1, cur ^ 1);
        const _Float16* kb = sm + cur * (2 * 64 * 72);
        const _Float16* vb = kb + 64 * 72;

        // S = Q @ K^T (16 x 64): 4 n-fragments, K-dim = 64 (2 steps)
        v8f sf[4];
#pragma unroll
        for (int nf = 0; nf < 4; ++nf) {
#pragma unroll
            for (int e = 0; e < 8; ++e) sf[nf][e] = 0.f;
            sf[nf] = wmma_f16(qa[0], frag16(kb + (nf * 16 + ml) * 72, g), sf[nf]);
            sf[nf] = wmma_f16(qa[1], frag16(kb + (nf * 16 + ml) * 72 + 32, g), sf[nf]);
        }

        // row max: 8 independent lane-tree reductions, step-outer so all 8
        // bpermutes issue per step (one dscnt wait per step, not per row)
        float mx[8];
#pragma unroll
        for (int r = 0; r < 8; ++r)
            mx[r] = fmaxf(fmaxf(sf[0][r], sf[1][r]), fmaxf(sf[2][r], sf[3][r]));
#pragma unroll
        for (int off = 1; off < 16; off <<= 1)
#pragma unroll
            for (int r = 0; r < 8; ++r)
                mx[r] = fmaxf(mx[r], shflx_f(mx[r], off));

        float nm[8], corr[8];
#pragma unroll
        for (int r = 0; r < 8; ++r) {
            nm[r] = fmaxf(mi[r], mx[r]);
            corr[r] = exp2_fast(mi[r] - nm[r]);
            mi[r] = nm[r];
        }
#pragma unroll
        for (int nf = 0; nf < 4; ++nf)
#pragma unroll
            for (int r = 0; r < 8; ++r)
                sf[nf][r] = exp2_fast(sf[nf][r] - nm[r]);
#pragma unroll
        for (int nf = 0; nf < 4; ++nf)
#pragma unroll
            for (int r = 0; r < 8; ++r) oacc[nf][r] *= corr[r];

        // bounce P through wave-private LDS: C-frag layout -> A-frag layout
#pragma unroll
        for (int nf = 0; nf < 4; ++nf)
#pragma unroll
            for (int r = 0; r < 8; ++r)
                pb[(r + 8 * g) * 72 + nf * 16 + ml] = (_Float16)sf[nf][r];
        asm volatile("s_wait_dscnt 0" ::: "memory");

        v16h pa[2];
#pragma unroll
        for (int kk = 0; kk < 2; ++kk) pa[kk] = frag16(pb + ml * 72 + kk * 32, g);

        // row sums via WMMA with all-ones B: every column of the result holds
        // the row sum, so each lane reads its rows' sums with no reduction.
        v8f lsum;
#pragma unroll
        for (int e = 0; e < 8; ++e) lsum[e] = 0.f;
        lsum = wmma_f16(pa[0], ones, lsum);
        lsum = wmma_f16(pa[1], ones, lsum);
#pragma unroll
        for (int r = 0; r < 8; ++r) li[r] = li[r] * corr[r] + lsum[r];

        // O += P @ V   (Vt rows = d, cols = s)
#pragma unroll
        for (int nf = 0; nf < 4; ++nf) {
            oacc[nf] = wmma_f16(pa[0], frag16(vb + (nf * 16 + ml) * 72, g), oacc[nf]);
            oacc[nf] = wmma_f16(pa[1], frag16(vb + (nf * 16 + ml) * 72 + 32, g), oacc[nf]);
        }

        wait_async0();
        __syncthreads();
    }

#pragma unroll
    for (int nf = 0; nf < 4; ++nf)
#pragma unroll
        for (int r = 0; r < 8; ++r) {
            float v = oacc[nf][r] / li[r];
            o[(size_t)(b * TT + t0 + r + 8 * g) * CC + h * DD + nf * 16 + ml] =
                (_Float16)v;
        }
}

// ---------------------------------------------------------------------------
extern "C" void kernel_launch(void* const* d_in, const int* in_sizes, int n_in,
                              void* d_out, int out_size, void* d_ws, size_t ws_size,
                              hipStream_t stream) {
    const float* x       = (const float*)d_in[0];
    const float* w_embed = (const float*)d_in[1];
    const float* b_embed = (const float*)d_in[2];
    const float* w_q     = (const float*)d_in[3];
    const float* w_k     = (const float*)d_in[4];
    const float* w_v     = (const float*)d_in[5];
    const float* w_proj  = (const float*)d_in[6];
    const float* b_proj  = (const float*)d_in[7];
    float* out = (float*)d_out;

    char* ws = (char*)d_ws;
    size_t off = 0;
    auto take = [&](size_t bytes) { char* p = ws + off; off += bytes; return p; };

    _Float16* xh     = (_Float16*)take((size_t)MM * INP * 2);
    _Float16* wembT  = (_Float16*)take((size_t)CC * INP * 2);
    _Float16* hbuf   = (_Float16*)take((size_t)MM * CC * 2);
    _Float16* wqT    = (_Float16*)take((size_t)CC * CC * 2);
    _Float16* wkT    = (_Float16*)take((size_t)CC * CC * 2);
    _Float16* wvT    = (_Float16*)take((size_t)CC * CC * 2);
    _Float16* wprojT = (_Float16*)take((size_t)ODP * CC * 2);
    _Float16* q_all  = (_Float16*)take((size_t)MM * CC * 2);
    _Float16* k_all  = (_Float16*)take((size_t)MM * CC * 2);
    _Float16* v_all  = (_Float16*)take((size_t)MM * CC * 2);
    _Float16* vT_all = (_Float16*)take((size_t)MM * CC * 2);
    _Float16* o_all  = (_Float16*)take((size_t)MM * CC * 2);
    (void)ws_size; (void)in_sizes; (void)n_in; (void)out_size;

    // prep
    prep_x<<<(MM * INP + 255) / 256, 256, 0, stream>>>(x, xh);
    prep_wembT<<<(CC * INP + 255) / 256, 256, 0, stream>>>(w_embed, wembT);
    prep_wqkv<<<(CC * CC + 255) / 256, 256, 0, stream>>>(w_q, w_k, w_v, wqT, wkT, wvT);
    prep_wprojT<<<(ODP * CC + 255) / 256, 256, 0, stream>>>(w_proj, wprojT);

    // h = x @ w_embed + b_embed          [8192,1024] f16
    gemm_f16w<true, true><<<dim3(MM / 128, CC / 128), 256, 0, stream>>>(
        xh, wembT, b_embed, hbuf, nullptr, INP, CC, CC, 1.0f);

    // q = (32*log2e) * h @ wq ; k = h @ wk ; v = h @ wv
    gemm_f16w<true, false><<<dim3(MM / 128, CC / 128), 256, 0, stream>>>(
        hbuf, wqT, nullptr, q_all, nullptr, CC, CC, CC, QSCALE);
    gemm_f16w<true, false><<<dim3(MM / 128, CC / 128), 256, 0, stream>>>(
        hbuf, wkT, nullptr, k_all, nullptr, CC, CC, CC, 1.0f);
    gemm_f16w<true, false><<<dim3(MM / 128, CC / 128), 256, 0, stream>>>(
        hbuf, wvT, nullptr, v_all, nullptr, CC, CC, CC, 1.0f);

    // v^T per (b,h)
    transpose_v<<<BB * HH * (TT / 64), 256, 0, stream>>>(v_all, vT_all);

    // flash attention -> o_all (concat-head layout)
    attn<<<BB * HH * (TT / 128), 256, 0, stream>>>(q_all, k_all, vT_all, o_all);

    // out = o_all @ w_proj + b_proj      [8192,136] f32
    gemm_f16w<false, true><<<dim3(MM / 128, ODP / 128), 256, 0, stream>>>(
        o_all, wprojT, b_proj, nullptr, out, CC, OD, OD, 1.0f);
}
